// AttentionDecoder_15925738734029
// MI455X (gfx1250) — matrix operand order repaired and checked by reading, not compiled, not run on hardware
//
#include <hip/hip_runtime.h>

typedef __attribute__((ext_vector_type(16))) __bf16 v16bf;
typedef __attribute__((ext_vector_type(8)))  __bf16 v8bf;
typedef __attribute__((ext_vector_type(8)))  float  v8f;

#define B_    32
#define SIZE_ 512
#define TDEC  64
#define TFEAT 512
#define VOCAB 10000
#define NPAD  10240           // VOCAB padded to a multiple of 64 for 4-tile N blocking
#define GDIM  1536            // 3*SIZE
#define ROWS  (TDEC * B_)     // 2048 batched time-major rows

// ---- fp32 -> bf16 round-to-nearest-even ----
__device__ __forceinline__ __bf16 f2bf(float f) {
  unsigned u = __float_as_uint(f);
  u += 0x7FFFu + ((u >> 16) & 1u);
  unsigned short s = (unsigned short)(u >> 16);
  union { unsigned short s; __bf16 b; } cv; cv.s = s; return cv.b;
}

// ---- register-blocked MTxNT 16x16-tile GEMM core over K (v_wmma_f32_16x16x32_bf16) ----
// A: activations at block origin, row-major [.., lda] bf16
// W: weights at block origin, row-major [.., ldw] bf16 (row n of W == column n of B)
// Per k-step: load MT A-fragments once, then stream one B-fragment per j and
// immediately consume it with MT WMMAs (keeps operand live-range small).
template <int MT, int NT>
__device__ __forceinline__ void gemm_block(const __bf16* __restrict__ A, int lda,
                                           const __bf16* __restrict__ W, int ldw,
                                           int K, v8f (&acc)[MT][NT]) {
  const int lane = threadIdx.x & 31;
  const int mn = lane & 15, h = lane >> 4;
  const __bf16* arow[MT];
  const __bf16* wrow[NT];
#pragma unroll
  for (int i = 0; i < MT; ++i) arow[i] = A + (size_t)(i * 16 + mn) * lda;
#pragma unroll
  for (int j = 0; j < NT; ++j) wrow[j] = W + (size_t)(j * 16 + mn) * ldw;
  for (int k0 = 0; k0 < K; k0 += 32) {
    union { v16bf v; v8bf c[2]; } a[MT];
#pragma unroll
    for (int i = 0; i < MT; ++i) {
      // A 16x32 layout: lane holds K in [8h,8h+8) and [16+8h,16+8h+8) of row m
      a[i].c[0] = *(const v8bf*)(arow[i] + k0 + 8 * h);
      a[i].c[1] = *(const v8bf*)(arow[i] + k0 + 16 + 8 * h);
    }
#pragma unroll
    for (int j = 0; j < NT; ++j) {
      union { v16bf v; v8bf c[2]; } b;
      // B 32x16 layout: lane holds K in [8h,8h+8) and [16+8h,16+8h+8) of column n
      b.c[0] = *(const v8bf*)(wrow[j] + k0 + 8 * h);
      b.c[1] = *(const v8bf*)(wrow[j] + k0 + 16 + 8 * h);
#pragma unroll
      for (int i = 0; i < MT; ++i)
        acc[i][j] = __builtin_amdgcn_wmma_f32_16x16x32_bf16(false, a[i].v, false, b.v,
                                                            (short)0, acc[i][j], false, false);
    }
  }
}

// ---- prologue: fp32 weight -> bf16 (with column slice for W_ih split) ----
__global__ void k_convert(const float* __restrict__ src, __bf16* __restrict__ dst,
                          int rows, int cols, int ld, int off) {
  int i = blockIdx.x * blockDim.x + threadIdx.x;
  if (i >= rows * cols) return;
  int r = i / cols, c = i - r * cols;
  dst[i] = f2bf(src[(size_t)r * ld + off + c]);
}

__global__ void k_zero_bf16(__bf16* dst, int n) {
  int i = blockIdx.x * blockDim.x + threadIdx.x;
  if (i < n) dst[i] = f2bf(0.f);
}

__global__ void k_init(float* h, float* ctx, __bf16* hb, __bf16* ctxb) {
  int i = blockIdx.x * blockDim.x + threadIdx.x;
  if (i < B_ * SIZE_) { h[i] = 0.f; ctx[i] = 0.f; hb[i] = f2bf(0.f); ctxb[i] = f2bf(0.f); }
}

// ---- batched embedding gather, time-major rows (row = t*32 + b) ----
__global__ void k_embed(const int* __restrict__ input, const float* __restrict__ emb,
                        __bf16* __restrict__ E) {
  int row = blockIdx.x;                 // 0..2047
  int t = row >> 5, b = row & 31;
  int tok = input[b * TDEC + t];        // padding row 0 already zero in emb
  const float* src = emb + (size_t)tok * SIZE_;
  __bf16* dst = E + (size_t)row * SIZE_;
  for (int d = threadIdx.x; d < SIZE_; d += blockDim.x) dst[d] = f2bf(src[d]);
}

// ---- batched: GX = E @ W_ih[:, :512]^T + b_ih  -> [2048, 1536] fp32 (4x4 blocked) ----
__global__ void __launch_bounds__(32, 1)
k_gemm_gxemb(const __bf16* __restrict__ E, const __bf16* __restrict__ WA,
             const float* __restrict__ b_ih, float* __restrict__ GX) {
  int nblk = blockIdx.x;   // 0..23  (1536/64)
  int mblk = blockIdx.y;   // 0..31  (2048/64)
  int lane = threadIdx.x & 31, mn = lane & 15, h = lane >> 4;
  v8f acc[4][4] = {};
  gemm_block<4, 4>(E + (size_t)mblk * 64 * SIZE_, SIZE_,
                   WA + (size_t)nblk * 64 * SIZE_, SIZE_, SIZE_, acc);
#pragma unroll
  for (int i = 0; i < 4; ++i) {
    int mbase = mblk * 64 + i * 16 + h * 8;
#pragma unroll
    for (int j = 0; j < 4; ++j) {
      int ncol = nblk * 64 + j * 16 + mn;
      float bias = b_ih[ncol];
#pragma unroll
      for (int r = 0; r < 8; ++r)
        GX[(size_t)(mbase + r) * GDIM + ncol] = acc[i][j][r] + bias;
    }
  }
}

// ---- per-step recurrent GEMMs (2x2 blocked, whole M=32 per wave):
//      gxc = ctx@W_ih[:,512:]^T ; gh = h@W_hh^T + b_hh ----
__global__ void __launch_bounds__(32, 1)
k_gemm_rec(const __bf16* __restrict__ ctxb, const __bf16* __restrict__ hb,
           const __bf16* __restrict__ WB, const __bf16* __restrict__ Whh,
           const float* __restrict__ b_hh,
           float* __restrict__ gxc, float* __restrict__ gh) {
  int nblk = blockIdx.x;                // 0..47 (1536/32)
  int which = blockIdx.y;               // 0 -> gxc, 1 -> gh
  int lane = threadIdx.x & 31, mn = lane & 15, h = lane >> 4;
  const __bf16* A = which ? hb : ctxb;
  const __bf16* W = which ? Whh : WB;
  float* Outp = which ? gh : gxc;
  v8f acc[2][2] = {};
  gemm_block<2, 2>(A, SIZE_, W + (size_t)nblk * 32 * SIZE_, SIZE_, SIZE_, acc);
#pragma unroll
  for (int i = 0; i < 2; ++i) {
    int mbase = i * 16 + h * 8;
#pragma unroll
    for (int j = 0; j < 2; ++j) {
      int ncol = nblk * 32 + j * 16 + mn;
      float bias = which ? b_hh[ncol] : 0.0f;
#pragma unroll
      for (int r = 0; r < 8; ++r)
        Outp[(size_t)(mbase + r) * GDIM + ncol] = acc[i][j][r] + bias;
    }
  }
}

// ---- GRU gate math (fp32), updates h, h_bf16, and the h-half of Y ----
__global__ void k_gates(const float* __restrict__ GX, const float* __restrict__ gxc,
                        const float* __restrict__ gh, float* __restrict__ h,
                        __bf16* __restrict__ hb, __bf16* __restrict__ Y, int t) {
  int i = blockIdx.x * blockDim.x + threadIdx.x;
  if (i >= B_ * SIZE_) return;
  int b = i >> 9, d = i & 511;
  const float* gxr = GX + (size_t)(t * B_ + b) * GDIM;
  const float* gxcr = gxc + (size_t)b * GDIM;
  const float* ghr = gh + (size_t)b * GDIM;
  float gx_r = gxr[d] + gxcr[d];
  float gx_z = gxr[d + 512] + gxcr[d + 512];
  float gx_n = gxr[d + 1024] + gxcr[d + 1024];
  float r = 1.f / (1.f + __expf(-(gx_r + ghr[d])));
  float z = 1.f / (1.f + __expf(-(gx_z + ghr[d + 512])));
  float n = tanhf(gx_n + r * ghr[d + 1024]);
  float hn = (1.f - z) * n + z * h[i];
  h[i] = hn;
  __bf16 hbf = f2bf(hn);
  hb[i] = hbf;
  Y[(size_t)(t * B_ + b) * 1024 + d] = hbf;
}

// ---- attention scores: sc[b,t'] = h[b,:] . features[b,t',:]  (grid 32x8, 64 thr) ----
__global__ void k_scores(const float* __restrict__ features, const float* __restrict__ h,
                         float* __restrict__ sc) {
  __shared__ float s_h[SIZE_];
  int b = blockIdx.x, chunk = blockIdx.y, tid = threadIdx.x;   // 64 threads
  for (int d = tid; d < SIZE_; d += 64) s_h[d] = h[b * SIZE_ + d];
  __syncthreads();
  int tt = chunk * 64 + tid;
  const float4* frow = (const float4*)(features + ((size_t)b * TFEAT + tt) * SIZE_);
  float acc = 0.f;
  for (int d4 = 0; d4 < SIZE_ / 4; ++d4) {
    float4 f4 = frow[d4];
    acc += f4.x * s_h[4 * d4] + f4.y * s_h[4 * d4 + 1] +
           f4.z * s_h[4 * d4 + 2] + f4.w * s_h[4 * d4 + 3];
  }
  sc[b * TFEAT + tt] = acc;
}

// ---- softmax over 512 scores; writes normalized weights output row ----
__global__ void k_softmax(const float* __restrict__ sc, float* __restrict__ wout, int t) {
  __shared__ float s_v[TFEAT];
  __shared__ float s_red[256];
  int b = blockIdx.x, tid = threadIdx.x;                       // 256 threads
  float m = -INFINITY;
  for (int i = tid; i < TFEAT; i += 256) { float v = sc[b * TFEAT + i]; s_v[i] = v; m = fmaxf(m, v); }
  s_red[tid] = m; __syncthreads();
  for (int s = 128; s > 0; s >>= 1) {
    if (tid < s) s_red[tid] = fmaxf(s_red[tid], s_red[tid + s]);
    __syncthreads();
  }
  float mx = s_red[0]; __syncthreads();
  float lsum = 0.f;
  for (int i = tid; i < TFEAT; i += 256) { float e = __expf(s_v[i] - mx); s_v[i] = e; lsum += e; }
  s_red[tid] = lsum; __syncthreads();
  for (int s = 128; s > 0; s >>= 1) {
    if (tid < s) s_red[tid] += s_red[tid + s];
    __syncthreads();
  }
  float inv = 1.f / s_red[0];
  float* wrow = wout + (size_t)(b * TDEC + t) * TFEAT;
  for (int i = tid; i < TFEAT; i += 256) wrow[i] = s_v[i] * inv;
}

// ---- context: ctx[b,d] = sum_t w[t]*features[b,t,d]  (grid 32x4, 128 thr) ----
__global__ void k_ctx(const float* __restrict__ features, const float* __restrict__ w,
                      float* __restrict__ ctx, __bf16* __restrict__ ctxb,
                      __bf16* __restrict__ Y, int t) {
  __shared__ float s_w[TFEAT];
  int b = blockIdx.x, dbase = blockIdx.y * 128, tid = threadIdx.x;   // 128 threads
  const float* wrow = w + (size_t)(b * TDEC + t) * TFEAT;
  for (int i = tid; i < TFEAT; i += 128) s_w[i] = wrow[i];
  __syncthreads();
  const float* fb = features + (size_t)b * TFEAT * SIZE_ + dbase + tid;
  float acc = 0.f;
  for (int i = 0; i < TFEAT; ++i) acc += s_w[i] * fb[(size_t)i * SIZE_];
  int d = dbase + tid;
  ctx[(size_t)b * SIZE_ + d] = acc;
  __bf16 cbf = f2bf(acc);
  ctxb[(size_t)b * SIZE_ + d] = cbf;
  Y[(size_t)(t * B_ + b) * 1024 + SIZE_ + d] = cbf;
}

// ---- deferred batched output projection (4x4 blocked): [2048,1024]@W_out^T + b_out ----
__global__ void __launch_bounds__(32, 1)
k_gemm_out(const __bf16* __restrict__ Y, const __bf16* __restrict__ Wo,
           const float* __restrict__ b_out, float* __restrict__ out) {
  int nblk = blockIdx.x;                // 0..159 (10240/64)
  int mblk = blockIdx.y;                // 0..31  (2048/64)
  int lane = threadIdx.x & 31, mn = lane & 15, h = lane >> 4;
  v8f acc[4][4] = {};
  gemm_block<4, 4>(Y + (size_t)mblk * 64 * 1024, 1024,
                   Wo + (size_t)nblk * 64 * 1024, 1024, 1024, acc);
#pragma unroll
  for (int i = 0; i < 4; ++i) {
    int mbase = mblk * 64 + i * 16 + h * 8;
#pragma unroll
    for (int j = 0; j < 4; ++j) {
      int ncol = nblk * 64 + j * 16 + mn;
      if (ncol < VOCAB) {
        float bias = b_out[ncol];
#pragma unroll
        for (int r = 0; r < 8; ++r) {
          int g = mbase + r;            // time-major row = t*32 + b
          int t = g >> 5, b = g & 31;
          out[(size_t)(b * TDEC + t) * VOCAB + ncol] = acc[i][j][r] + bias;
        }
      }
    }
  }
}

extern "C" void kernel_launch(void* const* d_in, const int* in_sizes, int n_in,
                              void* d_out, int out_size, void* d_ws, size_t ws_size,
                              hipStream_t stream) {
  (void)in_sizes; (void)n_in; (void)out_size; (void)ws_size;
  const int*   input    = (const int*)  d_in[0];
  const float* features = (const float*)d_in[1];
  // d_in[2] last_hidden is unused by the reference (h0 = zeros)
  const float* emb      = (const float*)d_in[3];
  const float* W_ih     = (const float*)d_in[4];
  const float* W_hh     = (const float*)d_in[5];
  const float* b_ih     = (const float*)d_in[6];
  const float* b_hh     = (const float*)d_in[7];
  const float* W_out    = (const float*)d_in[8];
  const float* b_out    = (const float*)d_in[9];

  float* out_logits = (float*)d_out;                                  // [32,64,10000]
  float* out_w = out_logits + (size_t)B_ * TDEC * VOCAB;              // [32,64,512]

  char* p = (char*)d_ws;
  auto carve = [&](size_t bytes) { char* r = p; p += (bytes + 255) & ~(size_t)255; return r; };
  __bf16* Wo_b  = (__bf16*)carve((size_t)NPAD * 1024 * 2);    // padded W_out bf16
  __bf16* WA_b  = (__bf16*)carve((size_t)GDIM * SIZE_ * 2);   // W_ih[:, :512]
  __bf16* WB_b  = (__bf16*)carve((size_t)GDIM * SIZE_ * 2);   // W_ih[:, 512:]
  __bf16* Whh_b = (__bf16*)carve((size_t)GDIM * SIZE_ * 2);
  __bf16* E_b   = (__bf16*)carve((size_t)ROWS * SIZE_ * 2);   // embeddings, all steps
  float*  GX    = (float*) carve((size_t)ROWS * GDIM * 4);    // gx(emb part)+b_ih, all steps
  float*  gxc   = (float*) carve((size_t)B_ * GDIM * 4);
  float*  ghb   = (float*) carve((size_t)B_ * GDIM * 4);
  float*  h     = (float*) carve((size_t)B_ * SIZE_ * 4);
  float*  ctx   = (float*) carve((size_t)B_ * SIZE_ * 4);
  float*  sc    = (float*) carve((size_t)B_ * TFEAT * 4);
  __bf16* h_b   = (__bf16*)carve((size_t)B_ * SIZE_ * 2);
  __bf16* ctx_b = (__bf16*)carve((size_t)B_ * SIZE_ * 2);
  __bf16* Y_b   = (__bf16*)carve((size_t)ROWS * 1024 * 2);    // [h,ctx] per step

  // prologue: weight conversion (+pad), state init, batched embed, batched gx GEMM
  k_convert<<<(VOCAB * 1024 + 255) / 256, 256, 0, stream>>>(W_out, Wo_b, VOCAB, 1024, 1024, 0);
  k_zero_bf16<<<((NPAD - VOCAB) * 1024 + 255) / 256, 256, 0, stream>>>(
      Wo_b + (size_t)VOCAB * 1024, (NPAD - VOCAB) * 1024);
  k_convert<<<(GDIM * SIZE_ + 255) / 256, 256, 0, stream>>>(W_ih, WA_b, GDIM, SIZE_, 1024, 0);
  k_convert<<<(GDIM * SIZE_ + 255) / 256, 256, 0, stream>>>(W_ih, WB_b, GDIM, SIZE_, 1024, SIZE_);
  k_convert<<<(GDIM * SIZE_ + 255) / 256, 256, 0, stream>>>(W_hh, Whh_b, GDIM, SIZE_, SIZE_, 0);
  k_init<<<(B_ * SIZE_ + 255) / 256, 256, 0, stream>>>(h, ctx, h_b, ctx_b);
  k_embed<<<ROWS, 256, 0, stream>>>(input, emb, E_b);
  k_gemm_gxemb<<<dim3(GDIM / 64, ROWS / 64), 32, 0, stream>>>(E_b, WA_b, b_ih, GX);

  // sequential recurrence
  for (int t = 0; t < TDEC; ++t) {
    k_gemm_rec<<<dim3(GDIM / 32, 2), 32, 0, stream>>>(ctx_b, h_b, WB_b, Whh_b, b_hh, gxc, ghb);
    k_gates<<<(B_ * SIZE_ + 255) / 256, 256, 0, stream>>>(GX, gxc, ghb, h, h_b, Y_b, t);
    k_scores<<<dim3(B_, TFEAT / 64), 64, 0, stream>>>(features, h, sc);
    k_softmax<<<B_, 256, 0, stream>>>(sc, out_w, t);
    k_ctx<<<dim3(B_, SIZE_ / 128), 128, 0, stream>>>(features, out_w, ctx, ctx_b, Y_b, t);
  }

  // epilogue: one big batched output projection (reads W_out once)
  k_gemm_out<<<dim3(NPAD / 64, ROWS / 64), 32, 0, stream>>>(Y_b, Wo_b, b_out, out_logits);
}
